// CausalSelfAttention_1365799600158
// MI455X (gfx1250) — matrix operand order repaired and checked by reading, not compiled, not run on hardware
//
#include <hip/hip_runtime.h>
#include <stdint.h>

// ---------------------------------------------------------------------------
// CDNA5 (gfx1250, wave32) causal GQA attention, bf16 WMMA everywhere.
// GEMMs: block-cooperative 128x128 tiles, async global->LDS double buffering.
// ---------------------------------------------------------------------------

#define WARP 32
#define T_SEQ 2048
#define C_DIM 2048
#define H_HEADS 16
#define G_KV 4
#define D_HEAD 128
#define N_QKV 3072           // (H + 2G) * D
#define TK (T_SEQ + 1)       // keys incl. sink
#define KROWS 2080           // padded key rows (multiple of 32 >= kb_last+32)
#define TKP 2080             // padded key columns for transposed V
#define LEFT_WIN 1022        // allowed: j >= i - 1022  (i.e. j >= i+1-1023)
#define SCALE_F 0.08838834764831845f  // 1/sqrt(128)
#define NEGM -1.0e30f

#define LDS_STR 40           // ushort stride of a staged 32-wide k-row (80 B)

typedef __attribute__((ext_vector_type(16))) __bf16 v16bf;
typedef __attribute__((ext_vector_type(8)))  float  v8f;

union V16BF { v16bf v; uint32_t u[8]; unsigned short h[16]; };
union V8F   { v8f v; float f[8]; };

__device__ __forceinline__ unsigned short f32_to_bf16(float f) {
  uint32_t x = __float_as_uint(f);
  uint32_t r = x + 0x7FFFu + ((x >> 16) & 1u);   // round-to-nearest-even
  return (unsigned short)(r >> 16);
}

// Build a 16-element bf16 A/B fragment from two 16-byte chunks.
__device__ __forceinline__ v16bf load_frag16(const unsigned short* p0,
                                             const unsigned short* p1) {
  V16BF t;
  uint4 a = *(const uint4*)p0;
  uint4 b = *(const uint4*)p1;
  t.u[0] = a.x; t.u[1] = a.y; t.u[2] = a.z; t.u[3] = a.w;
  t.u[4] = b.x; t.u[5] = b.y; t.u[6] = b.z; t.u[7] = b.w;
  return t.v;
}

// Async DMA of one 16-byte chunk: global -> LDS (tracked by ASYNCcnt).
// Low 32 bits of a generic pointer into LDS are the LDS byte address.
__device__ __forceinline__ void async_b128(unsigned short* lds_p,
                                           const unsigned short* g_p) {
  uint32_t lds_off = (uint32_t)(size_t)lds_p;
  asm volatile("global_load_async_to_lds_b128 %0, %1, off"
               :: "v"(lds_off), "v"(g_p)
               : "memory");
}

// ---------------------------------------------------------------------------
// 0) f32 -> bf16 conversion
// ---------------------------------------------------------------------------
__global__ void cvt_f32_to_bf16(const float* __restrict__ src,
                                unsigned short* __restrict__ dst, int n) {
  int i = blockIdx.x * blockDim.x + threadIdx.x;
  if (i < n) dst[i] = f32_to_bf16(src[i]);
}

// ---------------------------------------------------------------------------
// 1) GEMM: C[M][N] = A[M][K] * B[N][K]^T   (A,B bf16, C f32)
//    Block (256 thr / 8 waves) computes a 128x128 tile.
//    A/B k-chunks (128x32 bf16, padded rows) staged via async copy to LDS,
//    double buffered. Wave tile = 32x64 -> 8 WMMAs per k-chunk.
// ---------------------------------------------------------------------------
__global__ __launch_bounds__(256) void gemm_bf16_wmma(
    const unsigned short* __restrict__ A,
    const unsigned short* __restrict__ B,
    float* __restrict__ C, int M, int N, int K, int nbn) {
  __shared__ alignas(16) unsigned short sA[2][128 * LDS_STR];
  __shared__ alignas(16) unsigned short sB[2][128 * LDS_STR];

  int bm = blockIdx.x / nbn, bn = blockIdx.x % nbn;
  int m0 = bm * 128, n0 = bn * 128;
  int tid = threadIdx.x;
  int lane = tid & (WARP - 1);
  int w = tid >> 5;            // 0..7
  int wm = w & 3;              // 4 wave-rows of 32
  int wn = w >> 2;             // 2 wave-cols of 64
  int lo = lane & 15, hi = lane >> 4;

  // staging assignment: 512 16-byte chunks per tile, 2 per thread per matrix
  int r0 = (tid + 0) >> 2, p0 = (tid + 0) & 3;
  int r1 = (tid + 256) >> 2, p1 = (tid + 256) & 3;

  V8F acc[2][4];
  for (int s = 0; s < 2; s++)
    for (int j = 0; j < 4; j++)
      for (int r = 0; r < 8; r++) acc[s][j].f[r] = 0.0f;

  int nk = K / 32;
  // prologue: stage chunk 0 into buffer 0
  {
    async_b128(&sA[0][r0 * LDS_STR + p0 * 8], A + (size_t)(m0 + r0) * K + p0 * 8);
    async_b128(&sA[0][r1 * LDS_STR + p1 * 8], A + (size_t)(m0 + r1) * K + p1 * 8);
    async_b128(&sB[0][r0 * LDS_STR + p0 * 8], B + (size_t)(n0 + r0) * K + p0 * 8);
    async_b128(&sB[0][r1 * LDS_STR + p1 * 8], B + (size_t)(n0 + r1) * K + p1 * 8);
  }

  for (int kc = 0; kc < nk; kc++) {
    int cur = kc & 1;
    if (kc + 1 < nk) {
      int nxt = cur ^ 1;
      int k0 = (kc + 1) * 32;
      async_b128(&sA[nxt][r0 * LDS_STR + p0 * 8],
                 A + (size_t)(m0 + r0) * K + k0 + p0 * 8);
      async_b128(&sA[nxt][r1 * LDS_STR + p1 * 8],
                 A + (size_t)(m0 + r1) * K + k0 + p1 * 8);
      async_b128(&sB[nxt][r0 * LDS_STR + p0 * 8],
                 B + (size_t)(n0 + r0) * K + k0 + p0 * 8);
      async_b128(&sB[nxt][r1 * LDS_STR + p1 * 8],
                 B + (size_t)(n0 + r1) * K + k0 + p1 * 8);
      // 4 newest copies may remain in flight; the 4 older (chunk kc) complete
      // in order, so waiting <=4 means buffer `cur` is fully written.
      asm volatile("s_wait_asynccnt 0x4" ::: "memory");
    } else {
      asm volatile("s_wait_asynccnt 0x0" ::: "memory");
    }
    __syncthreads();   // make buffer `cur` visible block-wide

    // fragments out of LDS
    const unsigned short* pa0 = &sA[cur][(wm * 32 + lo) * LDS_STR + hi * 8];
    const unsigned short* pa1 = &sA[cur][(wm * 32 + 16 + lo) * LDS_STR + hi * 8];
    v16bf a0 = load_frag16(pa0, pa0 + 16);
    v16bf a1 = load_frag16(pa1, pa1 + 16);
    for (int j = 0; j < 4; j++) {
      const unsigned short* pb =
          &sB[cur][(wn * 64 + j * 16 + lo) * LDS_STR + hi * 8];
      v16bf bf = load_frag16(pb, pb + 16);
      acc[0][j].v = __builtin_amdgcn_wmma_f32_16x16x32_bf16(
          false, a0, false, bf, (short)0, acc[0][j].v, false, false);
      acc[1][j].v = __builtin_amdgcn_wmma_f32_16x16x32_bf16(
          false, a1, false, bf, (short)0, acc[1][j].v, false, false);
    }
    __syncthreads();   // protect buffer from the next overwrite
  }

  for (int s = 0; s < 2; s++)
    for (int j = 0; j < 4; j++)
      for (int r = 0; r < 8; r++)
        C[(size_t)(m0 + wm * 32 + s * 16 + r + hi * 8) * N +
          n0 + wn * 64 + j * 16 + lo] = acc[s][j].f[r];
}

// ---------------------------------------------------------------------------
// 2) RoPE + scatter: qkv f32 [T][3072] -> q bf16 [H][T][128],
//    k bf16 [G][KROWS][128] (sink at row 0), vT bf16 [G][128][TKP] (sink col 0)
// ---------------------------------------------------------------------------
__global__ void rope_scatter(const float* __restrict__ qkv,
                             const float* __restrict__ cosb,
                             const float* __restrict__ sinb,
                             const float* __restrict__ k_sink,
                             const float* __restrict__ v_sink,
                             unsigned short* __restrict__ qb,
                             unsigned short* __restrict__ kb,
                             unsigned short* __restrict__ vtb) {
  int idx = blockIdx.x * blockDim.x + threadIdx.x;
  if (idx >= T_SEQ * G_KV * 64) return;
  int d2 = idx & 63;
  int g = (idx >> 6) & 3;
  int t = idx >> 8;
  float c = cosb[t * 64 + d2], s = sinb[t * 64 + d2];
  const float* base = qkv + (size_t)t * N_QKV + g * 768;  // 6 * 128

  for (int sq = 0; sq < 4; sq++) {   // four query heads per kv group
    float lov = base[sq * 128 + d2];
    float hiv = base[sq * 128 + 64 + d2];
    int h = g * 4 + sq;
    unsigned short* qp = qb + ((size_t)h * T_SEQ + t) * D_HEAD;
    qp[d2] = f32_to_bf16(lov * c - hiv * s);
    qp[d2 + 64] = f32_to_bf16(lov * s + hiv * c);
  }
  {  // key (rope), shifted by 1 for the sink
    float lov = base[4 * 128 + d2];
    float hiv = base[4 * 128 + 64 + d2];
    unsigned short* kp = kb + ((size_t)g * KROWS + (t + 1)) * D_HEAD;
    kp[d2] = f32_to_bf16(lov * c - hiv * s);
    kp[d2 + 64] = f32_to_bf16(lov * s + hiv * c);
  }
  {  // value, transposed [d][key]
    float lov = base[5 * 128 + d2];
    float hiv = base[5 * 128 + 64 + d2];
    vtb[((size_t)g * D_HEAD + d2) * TKP + (t + 1)] = f32_to_bf16(lov);
    vtb[((size_t)g * D_HEAD + d2 + 64) * TKP + (t + 1)] = f32_to_bf16(hiv);
  }
  if (t == 0) {  // sink token at key index 0 (no rope on sink)
    kb[(size_t)g * KROWS * D_HEAD + d2] = f32_to_bf16(k_sink[g * 128 + d2]);
    kb[(size_t)g * KROWS * D_HEAD + d2 + 64] =
        f32_to_bf16(k_sink[g * 128 + d2 + 64]);
    vtb[((size_t)g * D_HEAD + d2) * TKP] = f32_to_bf16(v_sink[g * 128 + d2]);
    vtb[((size_t)g * D_HEAD + d2 + 64) * TKP] =
        f32_to_bf16(v_sink[g * 128 + d2 + 64]);
  }
}

// ---------------------------------------------------------------------------
// 3) Flash attention, one wave per (head, 16-query tile).
//    Scores via WMMA (Q x K^T), online softmax (shfl_xor row reductions),
//    P tile relaid via LDS into A-fragment layout, PV via WMMA with V^T.
// ---------------------------------------------------------------------------
__global__ __launch_bounds__(256) void attn_flash_wmma(
    const unsigned short* __restrict__ qb,
    const unsigned short* __restrict__ kbuf,
    const unsigned short* __restrict__ vtb,
    unsigned short* __restrict__ ob) {
  __shared__ alignas(16) unsigned short plds[8][16 * LDS_STR]; // per-wave P
  int lane = threadIdx.x & (WARP - 1);
  int wid = threadIdx.x >> 5;
  int wave = blockIdx.x * (blockDim.x / WARP) + wid;
  int head = wave >> 7;   // 128 q-tiles per head
  if (head >= H_HEADS) return;
  int qt = wave & 127;
  int q0 = qt * 16;
  int g = head >> 2;      // QPK = 4
  int lo = lane & 15, hi = lane >> 4;
  unsigned short* pl = plds[wid];

  // Q fragments for all four 32-wide K-chunks of D=128
  v16bf qa[4];
  {
    const unsigned short* qrow =
        qb + ((size_t)head * T_SEQ + (q0 + lo)) * D_HEAD + hi * 8;
    for (int c = 0; c < 4; c++)
      qa[c] = load_frag16(qrow + c * 32, qrow + c * 32 + 16);
  }

  float mrow[8], lrow[8];
  V8F oacc[8];
  for (int r = 0; r < 8; r++) { mrow[r] = NEGM; lrow[r] = 0.0f; }
  for (int n = 0; n < 8; n++)
    for (int r = 0; r < 8; r++) oacc[n].f[r] = 0.0f;

  int kb_start = q0 - LEFT_WIN;
  if (kb_start < 0) kb_start = 0;
  kb_start &= ~31;
  int kb_last = (q0 + 16) & ~31;

  for (int kb = kb_start; kb <= kb_last; kb += 32) {
    // ---- scores: two 16x16 tiles covering keys kb..kb+31 ----
    V8F s01[2];
    for (int jt = 0; jt < 2; jt++) {
      V8F sacc;
      for (int r = 0; r < 8; r++) sacc.f[r] = 0.0f;
      int n = kb + jt * 16 + lo;  // key column held by this lane
      const unsigned short* kp =
          kbuf + ((size_t)g * KROWS + n) * D_HEAD + hi * 8;
      for (int c = 0; c < 4; c++) {
        v16bf bf = load_frag16(kp + c * 32, kp + c * 32 + 16);
        sacc.v = __builtin_amdgcn_wmma_f32_16x16x32_bf16(
            false, qa[c], false, bf, (short)0, sacc.v, false, false);
      }
      for (int r = 0; r < 8; r++) {
        int i = q0 + r + hi * 8;
        bool ok = (n <= i + 1) && (n + LEFT_WIN >= i) && (n < TK);
        sacc.f[r] = ok ? sacc.f[r] * SCALE_F : NEGM;
      }
      s01[jt] = sacc;
    }
    // ---- online softmax update over 32 columns ----
    float pv0[8], pv1[8];
    for (int r = 0; r < 8; r++) {
      float mx = fmaxf(s01[0].f[r], s01[1].f[r]);
      mx = fmaxf(mx, __shfl_xor(mx, 8, 32));
      mx = fmaxf(mx, __shfl_xor(mx, 4, 32));
      mx = fmaxf(mx, __shfl_xor(mx, 2, 32));
      mx = fmaxf(mx, __shfl_xor(mx, 1, 32));
      float mnew = fmaxf(mrow[r], mx);
      float fac = __expf(mrow[r] - mnew);
      float p0 = __expf(s01[0].f[r] - mnew);
      float p1 = __expf(s01[1].f[r] - mnew);
      float rs = p0 + p1;
      rs += __shfl_xor(rs, 8, 32);
      rs += __shfl_xor(rs, 4, 32);
      rs += __shfl_xor(rs, 2, 32);
      rs += __shfl_xor(rs, 1, 32);
      lrow[r] = lrow[r] * fac + rs;
      mrow[r] = mnew;
      for (int nT = 0; nT < 8; nT++) oacc[nT].f[r] *= fac;
      pv0[r] = p0; pv1[r] = p1;
    }
    // ---- C-layout -> A-layout via LDS (bf16 16x32 tile, stride 40) ----
    for (int r = 0; r < 8; r++) {
      int row = r + hi * 8;
      pl[row * LDS_STR + lo] = f32_to_bf16(pv0[r]);
      pl[row * LDS_STR + 16 + lo] = f32_to_bf16(pv1[r]);
    }
    asm volatile("s_wait_dscnt 0" ::: "memory");
    v16bf pa;
    {
      const unsigned short* pr = pl + lo * LDS_STR + hi * 8;
      pa = load_frag16(pr, pr + 16);
    }
    // ---- PV: 8 WMMAs, B fragments straight out of transposed V ----
    for (int nT = 0; nT < 8; nT++) {
      const unsigned short* vp =
          vtb + ((size_t)g * D_HEAD + nT * 16 + lo) * TKP + kb + hi * 8;
      v16bf bf = load_frag16(vp, vp + 16);
      oacc[nT].v = __builtin_amdgcn_wmma_f32_16x16x32_bf16(
          false, pa, false, bf, (short)0, oacc[nT].v, false, false);
    }
  }

  // ---- epilogue: o = oacc / l, bf16 into [T][H*D] for the proj GEMM ----
  for (int r = 0; r < 8; r++) {
    float inv = 1.0f / lrow[r];
    int row = q0 + r + hi * 8;
    unsigned short* op = ob + (size_t)row * (H_HEADS * D_HEAD) + head * D_HEAD;
    for (int nT = 0; nT < 8; nT++)
      op[nT * 16 + lo] = f32_to_bf16(oacc[nT].f[r] * inv);
  }
}

// ---------------------------------------------------------------------------
// launch
// ---------------------------------------------------------------------------
extern "C" void kernel_launch(void* const* d_in, const int* in_sizes, int n_in,
                              void* d_out, int out_size, void* d_ws,
                              size_t ws_size, hipStream_t stream) {
  const float* x = (const float*)d_in[0];        // [T][C]
  const float* cosb = (const float*)d_in[1];     // [T][64]
  const float* sinb = (const float*)d_in[2];     // [T][64]
  const float* W_attn = (const float*)d_in[3];   // [3072][2048]
  const float* W_proj = (const float*)d_in[4];   // [2048][2048]
  const float* k_sink = (const float*)d_in[5];   // [512]
  const float* v_sink = (const float*)d_in[6];   // [512]
  float* y = (float*)d_out;                      // [T][C]

  char* ws = (char*)d_ws;
  size_t off = 0;
  auto wsalloc = [&](size_t bytes) -> void* {
    void* p = ws + off;
    off += (bytes + 255) & ~(size_t)255;
    return p;
  };
  unsigned short* xb = (unsigned short*)wsalloc((size_t)T_SEQ * C_DIM * 2);
  unsigned short* wab = (unsigned short*)wsalloc((size_t)N_QKV * C_DIM * 2);
  unsigned short* wpb = (unsigned short*)wsalloc((size_t)C_DIM * C_DIM * 2);
  float* qkv = (float*)wsalloc((size_t)T_SEQ * N_QKV * 4);
  unsigned short* qb =
      (unsigned short*)wsalloc((size_t)H_HEADS * T_SEQ * D_HEAD * 2);
  unsigned short* kb =
      (unsigned short*)wsalloc((size_t)G_KV * KROWS * D_HEAD * 2);
  unsigned short* vtb =
      (unsigned short*)wsalloc((size_t)G_KV * D_HEAD * TKP * 2);
  unsigned short* ob = (unsigned short*)wsalloc((size_t)T_SEQ * C_DIM * 2);

  // 0) convert inputs to bf16
  {
    int n = T_SEQ * C_DIM;
    cvt_f32_to_bf16<<<(n + 255) / 256, 256, 0, stream>>>(x, xb, n);
  }
  {
    int n = N_QKV * C_DIM;
    cvt_f32_to_bf16<<<(n + 255) / 256, 256, 0, stream>>>(W_attn, wab, n);
  }
  {
    int n = C_DIM * C_DIM;
    cvt_f32_to_bf16<<<(n + 255) / 256, 256, 0, stream>>>(W_proj, wpb, n);
  }

  // 1) qkv = x @ W_attn^T   (M=2048, N=3072, K=2048): 16x24 = 384 blocks
  {
    int nbm = T_SEQ / 128, nbn = N_QKV / 128;
    gemm_bf16_wmma<<<nbm * nbn, 256, 0, stream>>>(xb, wab, qkv, T_SEQ, N_QKV,
                                                  C_DIM, nbn);
  }

  // 2) RoPE + scatter (+ sink fill, + V transpose)
  {
    int n = T_SEQ * G_KV * 64;   // 524288
    rope_scatter<<<n / 256, 256, 0, stream>>>(qkv, cosb, sinb, k_sink, v_sink,
                                              qb, kb, vtb);
  }

  // 3) flash attention: 2048 waves (16 heads x 128 q-tiles)
  {
    int waves = H_HEADS * (T_SEQ / 16);
    attn_flash_wmma<<<waves / 8, 256, 0, stream>>>(qb, kb, vtb, ob);
  }

  // 4) y = o @ W_proj^T   (M=2048, N=2048, K=2048): 16x16 = 256 blocks
  {
    int nbm = T_SEQ / 128, nbn = C_DIM / 128;
    gemm_bf16_wmma<<<nbm * nbn, 256, 0, stream>>>(ob, wpb, y, T_SEQ, C_DIM,
                                                  C_DIM, nbn);
  }
}
